// FieldsWithInteractionModuleAndLSTM_37271726195514
// MI455X (gfx1250) — compile-verified, matrix operand order checked by reading, not run on hardware
//
#include <hip/hip_runtime.h>
#include <hip/hip_bf16.h>

typedef __attribute__((ext_vector_type(16))) _Float16 v16h;
typedef __attribute__((ext_vector_type(8)))  _Float16 v8h;
typedef __attribute__((ext_vector_type(8)))  float    v8f;

constexpr int NTOT = 131072;
constexpr int NSEQ = 8192;
constexpr int HDIM = 128;
constexpr int NKF  = 20;
constexpr int PRED = 12;
constexpr int MTILE = 64;     // rows per workgroup
constexpr int KX    = 384;    // [emb(128) | pool(128) | h(128)]
constexpr int GOUT  = 512;    // 4*H gates

// ---- order-preserving float<->u32 encoding for atomic max --------------------
__device__ __forceinline__ unsigned enc_u(float f) {
    unsigned u = __float_as_uint(f);
    return (u & 0x80000000u) ? ~u : (u | 0x80000000u);
}
__device__ __forceinline__ float dec_u(unsigned u) {
    u = (u & 0x80000000u) ? (u & 0x7FFFFFFFu) : ~u;
    return __uint_as_float(u);
}
__device__ __forceinline__ float sigf(float x) { return 1.0f / (1.0f + __expf(-x)); }

// ---- A-fragment (16x32 f16) from LDS, ISA §7.12.2 layout ---------------------
// lane l holds row m=l%16; element e -> K = kbase + (e%8) + 8*(l>=16) + 16*(e>=8)
// => two contiguous 16B chunks at kbase+(l>=16?8:0) and +16 elements further.
__device__ __forceinline__ v16h load_frag_a(const _Float16* base, int stride, int m, int k0) {
    const _Float16* p = base + m * stride + k0;
    v8h lo = *(const v8h*)(p);
    v8h hi = *(const v8h*)(p + 16);
    v16h r;
#pragma unroll
    for (int e = 0; e < 8; ++e) { r[e] = lo[e]; r[e + 8] = hi[e]; }
    return r;
}

// ---- one-time weight swizzle: pack B fragments contiguous per (tile,kchunk,lane)
__global__ void swizzle_weights(const float* __restrict__ Wx, const float* __restrict__ Wh,
                                const float* __restrict__ Wpool,
                                _Float16* __restrict__ WcatB, _Float16* __restrict__ WpoolB) {
    int idx = blockIdx.x * 256 + threadIdx.x;
    const int NG = 32 * 12 * 512;          // gates: 32 n-tiles x 12 k-chunks
    if (idx < NG) {
        int fi = idx >> 9, r = idx & 511;
        int lane = r >> 4, e = r & 15;
        int nt = fi / 12, kc = fi % 12;
        int k = kc * 32 + (e & 7) + ((lane & 16) ? 8 : 0) + ((e >> 3) ? 16 : 0);
        int n = nt * 16 + (lane & 15);
        float v = (k < 256) ? Wx[k * GOUT + n] : Wh[(k - 256) * GOUT + n];
        WcatB[fi * 512 + lane * 16 + e] = (_Float16)v;
    } else if (idx < NG + 8 * 4 * 512) {   // pool: 8 n-tiles x 4 k-chunks
        int i2 = idx - NG;
        int fi = i2 >> 9, r = i2 & 511;
        int lane = r >> 4, e = r & 15;
        int np = fi >> 2, kc = fi & 3;
        int k = kc * 32 + (e & 7) + ((lane & 16) ? 8 : 0) + ((e >> 3) ? 16 : 0);
        int n = np * 16 + (lane & 15);
        WpoolB[fi * 512 + lane * 16 + e] = (_Float16)Wpool[k * HDIM + n];
    }
}

// ---- init: h=state_final, c=0 ----------------------------------------------
__global__ void copy_hc(const float* __restrict__ sf, float* __restrict__ Hs, float* __restrict__ Cs) {
    int i = blockIdx.x * 256 + threadIdx.x;
    if (i < NTOT * HDIM) { Hs[i] = sf[i]; Cs[i] = 0.0f; }
}

// ---- init: out0/rel/pos/alpha-norm/fd --------------------------------------
__global__ void rowinit(const float* __restrict__ sf, const float* __restrict__ lastpos,
                        const float* __restrict__ alpha0, const float* __restrict__ fd0,
                        const float* __restrict__ Wout, const float* __restrict__ bout,
                        float* __restrict__ Rel, float* __restrict__ Pos,
                        float* __restrict__ Alp, float* __restrict__ Fd, float* __restrict__ Out) {
    int row = blockIdx.x * 256 + threadIdx.x;
    if (row >= NTOT) return;
    float o0 = bout[0], o1 = bout[1];
    for (int j = 0; j < HDIM; ++j) {
        float h = sf[row * HDIM + j];
        o0 += h * Wout[j * 2]; o1 += h * Wout[j * 2 + 1];
    }
    Rel[row * 2] = o0; Rel[row * 2 + 1] = o1;
    Pos[row * 2] = lastpos[row * 2] + o0; Pos[row * 2 + 1] = lastpos[row * 2 + 1] + o1;
    Out[row * 2] = o0; Out[row * 2 + 1] = o1;
    float s = 0.0f;
    for (int k = 0; k < NKF; ++k) s += alpha0[row * NKF + k];
    float inv = 1.0f / s;
    for (int k = 0; k < NKF; ++k) Alp[row * NKF + k] = alpha0[row * NKF + k] * inv;
    for (int t = 0; t < 2 * NKF; ++t) Fd[row * 2 * NKF + t] = fd0[row * 2 * NKF + t];
}

// ---- segment max (encoded u32 atomics; exact & deterministic) ---------------
__global__ void segmax_clear(unsigned* __restrict__ segmax) {
    int i = blockIdx.x * 256 + threadIdx.x;
    if (i < NSEQ * HDIM) segmax[i] = 0u;   // below enc() of any finite float from a real row
}
__global__ void segmax_atomic(const float* __restrict__ Hs, const int* __restrict__ segid,
                              unsigned* __restrict__ segmax) {
    int i = blockIdx.x * 256 + threadIdx.x;
    if (i < NTOT * HDIM) {
        int row = i >> 7, c = i & 127;
        atomicMax(&segmax[segid[row] * HDIM + c], enc_u(Hs[i]));
    }
}

// ---- fused per-step kernel ---------------------------------------------------
__global__ __launch_bounds__(256)
void step_kernel(const float* __restrict__ Wpos, const float* __restrict__ bpos,
                 const float* __restrict__ Wfld, const float* __restrict__ bfld,
                 const float* __restrict__ bpool, const float* __restrict__ blstm,
                 const float* __restrict__ Wout, const float* __restrict__ bout,
                 const float* __restrict__ fieldA, const float* __restrict__ Trans,
                 const int* __restrict__ segid, const unsigned char* __restrict__ mask,
                 const _Float16* __restrict__ WcatB, const _Float16* __restrict__ WpoolB,
                 const unsigned* __restrict__ segmax,
                 float* __restrict__ Hs, float* __restrict__ Cs,
                 float* __restrict__ Rel, float* __restrict__ Pos,
                 float* __restrict__ Alp, float* __restrict__ Fd,
                 float* __restrict__ Out, int step) {
    __shared__ alignas(16) _Float16 xs[MTILE * KX];     // [emb|pool|h] fp16, 48KB
    __shared__ alignas(16) _Float16 ps[MTILE * HDIM];   // gathered seg-max pool, 16KB
    __shared__ float h2s[MTILE * HDIM];                 // new h (f32) for head, 32KB
    __shared__ float dtp[MTILE * 2 * NKF];              // fields*alpha, 10KB

    const int tid = threadIdx.x;
    const int lane = tid & 31;
    const int wave = tid >> 5;
    const int row0 = blockIdx.x * MTILE;
    const int half = (lane >> 4) & 1;
    const int kb = half ? 8 : 0;

    // Phase 0: dtp + pool gather + h -> LDS
    for (int i = tid; i < MTILE * 2 * NKF; i += 256) {
        int r = i / (2 * NKF), t = i % (2 * NKF);
        int row = row0 + r;
        dtp[i] = Fd[row * 2 * NKF + t] * Alp[row * NKF + (t >> 1)];
    }
    for (int i = tid; i < MTILE * HDIM; i += 256) {
        int r = i >> 7, c = i & 127;
        int row = row0 + r;
        ps[r * HDIM + c] = (_Float16)dec_u(segmax[segid[row] * HDIM + c]);
        xs[r * KX + 256 + c] = (_Float16)Hs[row * HDIM + c];
    }
    __syncthreads();

    // Phase 1: pos_emb (cols 0..63) + fld_emb (cols 64..127)
    for (int i = tid; i < MTILE * 64; i += 256) {
        int r = i >> 6, j = i & 63;
        int row = row0 + r;
        float r0 = Rel[row * 2], r1 = Rel[row * 2 + 1];
        float pe = r0 * Wpos[j] + r1 * Wpos[64 + j] + bpos[j];
        float fe = bfld[j];
#pragma unroll 8
        for (int t = 0; t < 2 * NKF; ++t) fe += dtp[r * 2 * NKF + t] * Wfld[t * 64 + j];
        xs[r * KX + j]      = (_Float16)pe;
        xs[r * KX + 64 + j] = (_Float16)fe;
    }
    __syncthreads();

    // Phase 2: pool GEMM (64x128x128) via WMMA, relu, fp16 -> x cols 128..255
    const v16h* WPB = (const v16h*)WpoolB;
    for (int s = 0; s < 4; ++s) {
        int id = wave * 4 + s;              // 32 tile-tasks
        int rt = id >> 3, np = id & 7;
        v8f acc = {};
        int m = rt * 16 + (lane & 15);
#pragma unroll
        for (int kc = 0; kc < 4; ++kc) {
            v16h a = load_frag_a(ps, HDIM, m, kc * 32 + kb);
            v16h b = WPB[(np * 4 + kc) * 32 + lane];
            acc = __builtin_amdgcn_wmma_f32_16x16x32_f16(false, a, false, b, (short)0, acc, false, false);
        }
        int n = np * 16 + (lane & 15);
        float bia = bpool[n];
#pragma unroll
        for (int v = 0; v < 8; ++v) {
            float val = acc[v] + bia;
            val = val > 0.0f ? val : 0.0f;
            int lm = rt * 16 + v + 8 * half;
            xs[lm * KX + 128 + n] = (_Float16)val;
        }
    }
    __syncthreads();

    // Phase 3: gates GEMM (64x384x512). Each wave owns (row-tile, h-col group):
    // i/f/g/o tiles for the same h columns stay in registers -> LSTM in VGPRs.
    const v16h* WB = (const v16h*)WcatB;
    for (int s = 0; s < 4; ++s) {
        int id = wave * 4 + s;
        int rt = id >> 3, cg = id & 7;
        v8f ai = {}, af = {}, ag = {}, ao = {};
        int m = rt * 16 + (lane & 15);
#pragma unroll
        for (int kc = 0; kc < 12; ++kc) {
            v16h a  = load_frag_a(xs, KX, m, kc * 32 + kb);
            v16h bi = WB[((cg)      * 12 + kc) * 32 + lane];
            v16h bf = WB[((cg +  8) * 12 + kc) * 32 + lane];
            v16h bg = WB[((cg + 16) * 12 + kc) * 32 + lane];
            v16h bo = WB[((cg + 24) * 12 + kc) * 32 + lane];
            ai = __builtin_amdgcn_wmma_f32_16x16x32_f16(false, a, false, bi, (short)0, ai, false, false);
            af = __builtin_amdgcn_wmma_f32_16x16x32_f16(false, a, false, bf, (short)0, af, false, false);
            ag = __builtin_amdgcn_wmma_f32_16x16x32_f16(false, a, false, bg, (short)0, ag, false, false);
            ao = __builtin_amdgcn_wmma_f32_16x16x32_f16(false, a, false, bo, (short)0, ao, false, false);
        }
        int hcol = cg * 16 + (lane & 15);
        float b_i = blstm[hcol], b_f = blstm[128 + hcol];
        float b_g = blstm[256 + hcol], b_o = blstm[384 + hcol];
#pragma unroll
        for (int v = 0; v < 8; ++v) {
            int lm = rt * 16 + v + 8 * half;
            int row = row0 + lm;
            float gi = ai[v] + b_i, gf = af[v] + b_f;
            float gg = ag[v] + b_g, go = ao[v] + b_o;
            float co = Cs[row * HDIM + hcol], ho = Hs[row * HDIM + hcol];
            float cn = sigf(gf) * co + sigf(gi) * tanhf(gg);
            float hn = sigf(go) * tanhf(cn);
            bool mk = mask[row] != 0;
            float c2 = mk ? cn : co;
            float h2 = mk ? hn : ho;
            Cs[row * HDIM + hcol] = c2;
            Hs[row * HDIM + hcol] = h2;
            h2s[lm * HDIM + hcol] = h2;
        }
    }
    __syncthreads();

    // Phase 4: per-row head + state propagation
    if (tid < MTILE) {
        int lm = tid, row = row0 + lm;
        float o0 = bout[0], o1 = bout[1];
        for (int j = 0; j < HDIM; ++j) {
            float hv = h2s[lm * HDIM + j];
            o0 += hv * Wout[j * 2]; o1 += hv * Wout[j * 2 + 1];
        }
        bool mk = mask[row] != 0;
        float rn0 = mk ? o0 : Rel[row * 2];
        float rn1 = mk ? o1 : Rel[row * 2 + 1];
        float pn0 = Pos[row * 2] + rn0, pn1 = Pos[row * 2 + 1] + rn1;
        Rel[row * 2] = rn0; Rel[row * 2 + 1] = rn1;
        Pos[row * 2] = pn0; Pos[row * 2 + 1] = pn1;
        Out[(size_t)step * NTOT * 2 + row * 2]     = rn0;
        Out[(size_t)step * NTOT * 2 + row * 2 + 1] = rn1;
        // alpha @ trans -> softmax
        float al[NKF], lg[NKF];
        for (int k = 0; k < NKF; ++k) al[k] = Alp[row * NKF + k];
        float mx = -1e30f;
        for (int j = 0; j < NKF; ++j) {
            float sm = 0.0f;
            for (int i = 0; i < NKF; ++i) sm += al[i] * Trans[i * NKF + j];
            lg[j] = sm; mx = fmaxf(mx, sm);
        }
        float den = 0.0f;
        for (int j = 0; j < NKF; ++j) { lg[j] = __expf(lg[j] - mx); den += lg[j]; }
        float inv = 1.0f / den;
        if (mk) {
            for (int j = 0; j < NKF; ++j) Alp[row * NKF + j] = lg[j] * inv;
            for (int k = 0; k < NKF; ++k) {
                float f0 = pn0 * fieldA[k * 4 + 0] + pn1 * fieldA[k * 4 + 2];
                float f1 = pn0 * fieldA[k * 4 + 1] + pn1 * fieldA[k * 4 + 3];
                Fd[row * 2 * NKF + 2 * k]     = f0;
                Fd[row * 2 * NKF + 2 * k + 1] = f1;
            }
        }
    }
}

extern "C" void kernel_launch(void* const* d_in, const int* in_sizes, int n_in,
                              void* d_out, int out_size, void* d_ws, size_t ws_size,
                              hipStream_t stream) {
    const float* sf      = (const float*)d_in[0];
    const float* lastpos = (const float*)d_in[1];
    const float* alpha0  = (const float*)d_in[2];
    const float* fd0     = (const float*)d_in[3];
    const float* Wpos    = (const float*)d_in[4];
    const float* bpos    = (const float*)d_in[5];
    const float* Wfld    = (const float*)d_in[6];
    const float* bfld    = (const float*)d_in[7];
    const float* Wpool   = (const float*)d_in[8];
    const float* bpool   = (const float*)d_in[9];
    const float* Wx      = (const float*)d_in[10];
    const float* Wh      = (const float*)d_in[11];
    const float* blstm   = (const float*)d_in[12];
    const float* Wout    = (const float*)d_in[13];
    const float* bout    = (const float*)d_in[14];
    const float* fieldA  = (const float*)d_in[15];
    const float* Trans   = (const float*)d_in[16];
    const int*   segid   = (const int*)d_in[17];
    const unsigned char* mask = (const unsigned char*)d_in[18];
    float* Out = (float*)d_out;

    char* ws = (char*)d_ws;
    size_t off = 0;
    auto take = [&](size_t bytes) -> char* {
        char* p = ws + off;
        off += (bytes + 255) & ~(size_t)255;
        return p;
    };
    float*    Hs     = (float*)take((size_t)NTOT * HDIM * 4);
    float*    Cs     = (float*)take((size_t)NTOT * HDIM * 4);
    float*    Rel    = (float*)take((size_t)NTOT * 2 * 4);
    float*    Pos    = (float*)take((size_t)NTOT * 2 * 4);
    float*    Alp    = (float*)take((size_t)NTOT * NKF * 4);
    float*    Fd     = (float*)take((size_t)NTOT * 2 * NKF * 4);
    unsigned* Smax   = (unsigned*)take((size_t)NSEQ * HDIM * 4);
    _Float16* WcatB  = (_Float16*)take((size_t)KX * GOUT * 2);
    _Float16* WpoolB = (_Float16*)take((size_t)HDIM * HDIM * 2);

    swizzle_weights<<<(32 * 12 * 512 + 8 * 4 * 512 + 255) / 256, 256, 0, stream>>>(
        Wx, Wh, Wpool, WcatB, WpoolB);
    copy_hc<<<(NTOT * HDIM) / 256, 256, 0, stream>>>(sf, Hs, Cs);
    rowinit<<<NTOT / 256, 256, 0, stream>>>(sf, lastpos, alpha0, fd0, Wout, bout,
                                            Rel, Pos, Alp, Fd, Out);
    for (int t = 1; t < PRED; ++t) {
        segmax_clear<<<(NSEQ * HDIM) / 256, 256, 0, stream>>>(Smax);
        segmax_atomic<<<(NTOT * HDIM) / 256, 256, 0, stream>>>(Hs, segid, Smax);
        step_kernel<<<NTOT / MTILE, 256, 0, stream>>>(
            Wpos, bpos, Wfld, bfld, bpool, blstm, Wout, bout, fieldA, Trans,
            segid, mask, WcatB, WpoolB, Smax, Hs, Cs, Rel, Pos, Alp, Fd, Out, t);
    }
    (void)in_sizes; (void)n_in; (void)out_size; (void)ws_size;
}
